// SpiralAutoencoder_90572270338577
// MI455X (gfx1250) — matrix operand order repaired and checked by reading, not compiled, not run on hardware
//
#include <hip/hip_runtime.h>
#include <cstdint>

// ---------------------------------------------------------------------------
// SpiralNet mesh autoencoder forward on MI455X (gfx1250, wave32, WMMA).
// Generic bf16-WMMA GEMM (f32 accumulate) for spiral-conv (fused gather),
// dense pool, and FC layers. LDS tiles are stored PRE-PERMUTED into the CDNA5
// WMMA 16-bit fragment layout (fragment fetch = 2x ds_load_b128 per lane) and
// DOUBLE-BUFFERED so the next stage's global loads overlap this stage's WMMAs
// with a single workgroup barrier per K-stage.
// ---------------------------------------------------------------------------

typedef __bf16 bf16_t;
typedef __attribute__((ext_vector_type(16))) __bf16 v16bf;
typedef __attribute__((ext_vector_type(8)))  __bf16 v8bf;
typedef __attribute__((ext_vector_type(8)))  float  v8f;

#define BLOCK_THREADS 128   // 4 waves of 32
#define WAVES 4
#define TM 64               // rows per block (16 per wave)
#define TN 32               // output cols per block (2 n-tiles)
#define TK 64               // K per stage (2 WMMA k-steps of 32)

// A 16x32 bf16 fragment layout (lane l, vgpr v):
//   kb = ((v<4) ? 2v : 16+2(v-4)) + 8*(l>>4); elements K = kb, kb+1; row M = l&15.
// An 8-aligned group of 8 K-columns of one row maps to 8 consecutive bf16
// slots of a single lane. B 32x16: row r -> lane ((r>>3)&1)*16 + col, elem (r&7)+8*(r>>4)
// (same 8-group property along K).

// flags: bit0 = ELU activation, bit1 = zero last row (pad vertex)
template <int FI>
__global__ __launch_bounds__(BLOCK_THREADS)
void spiral_gemm_bf16_wmma(
    int M, int Kdim, int Ndim,
    const float* __restrict__ A, long long Abs, int lda,     // FI==0 path
    const long long* __restrict__ gather,                    // [M][9] spiral idx
    const float* __restrict__ X, long long Xbs,              // FI>0 path
    const float* __restrict__ Bm, long long Bbs, int ldb,
    float* __restrict__ C, long long Cbs, int ldc,
    const float* __restrict__ bias, int flags)
{
    // Double-buffered LDS tiles in fragment order: [buf][.][ksub][lane][elem]
    __shared__ __align__(16) bf16_t Asd[2][WAVES][2][32][16];   // 16 KB
    __shared__ __align__(16) bf16_t Bsd[2][2][2][32][16];       //  8 KB

    const int tid  = threadIdx.x;
    const int lane = tid & 31;
    const int wave = tid >> 5;
    const int bz   = blockIdx.z;
    const int m0   = blockIdx.y * TM;
    const int n0   = blockIdx.x * TN;

    const long long aOff = (long long)bz * Abs;
    const long long xOff = (long long)bz * Xbs;
    const long long bOff = (long long)bz * Bbs;
    const long long cOff = (long long)bz * Cbs;

    v8f acc0 = {};
    v8f acc1 = {};

    // ---- stage one TK-slab of A and B into LDS buffer `sbuf` ----
    auto stage = [&](int sbuf, int kk) {
        // A tile (TM x TK), groups of 8 columns -> one ds_store_b128
        for (int g = tid; g < TM * (TK / 8); g += BLOCK_THREADS) {
            const int r  = g >> 3;            // tile row 0..63
            const int cg = (g & 7) << 3;      // col base 0,8,..,56
            const int m  = m0 + r;
            const int kc = kk + cg;
            float vals[8];
            if constexpr (FI == 0) {
                if (m < M && kc + 8 <= Kdim) {          // fast interior path
                    const float* p = A + aOff + (long long)m * lda + kc;
#pragma unroll
                    for (int j = 0; j < 8; ++j) vals[j] = p[j];
                } else {
#pragma unroll
                    for (int j = 0; j < 8; ++j) {
                        const int k = kc + j;
                        vals[j] = (m < M && k < Kdim)
                                      ? A[aOff + (long long)m * lda + k] : 0.0f;
                    }
                }
            } else if constexpr (FI == 3) {
#pragma unroll
                for (int j = 0; j < 8; ++j) {
                    const int k = kc + j;
                    float v = 0.0f;
                    if (m < M && k < Kdim) {
                        const int ks = k / 3;
                        const int fi = k - ks * 3;
                        v = X[xOff + gather[(long long)m * 9 + ks] * 3 + fi];
                    }
                    vals[j] = v;
                }
            } else {
                // FI = pow2 >= 64: whole group inside one spiral slot.
                // base = vtx*FI + fi is a multiple of 8 floats -> float4 ok.
                const bool ok = (m < M) && (kc < Kdim);
                if (ok) {
                    const int ks = kc / FI;              // compile-time shift
                    const int fi = kc & (FI - 1);
                    const long long base =
                        xOff + gather[(long long)m * 9 + ks] * (long long)FI + fi;
                    const float4 f0 = *(const float4*)(X + base);
                    const float4 f1 = *(const float4*)(X + base + 4);
                    vals[0] = f0.x; vals[1] = f0.y; vals[2] = f0.z; vals[3] = f0.w;
                    vals[4] = f1.x; vals[5] = f1.y; vals[6] = f1.z; vals[7] = f1.w;
                } else {
#pragma unroll
                    for (int j = 0; j < 8; ++j) vals[j] = 0.0f;
                }
            }
            const int s      = cg >> 5;               // ksub 0/1
            const int c32    = cg & 31;
            const int lane_d = (((c32 >> 3) & 1) << 4) + (r & 15);
            const int ebase  = ((c32 >> 4) & 1) << 3;
            v8bf pk;
#pragma unroll
            for (int j = 0; j < 8; ++j) pk[j] = (bf16_t)vals[j];
            *(v8bf*)&Asd[sbuf][r >> 4][s][lane_d][ebase] = pk;
        }

        // B tile (TK x TN), groups of 8 k-rows -> one ds_store_b128
        for (int g = tid; g < (TK / 8) * TN; g += BLOCK_THREADS) {
            const int c  = g & 31;            // col 0..31
            const int rb = (g >> 5) << 3;     // k-row base 0,8,..,56
            const int n  = n0 + c;
            float vals[8];
            if (n < Ndim && kk + rb + 8 <= Kdim) {      // fast interior path
                const float* p = Bm + bOff + (long long)(kk + rb) * ldb + n;
#pragma unroll
                for (int j = 0; j < 8; ++j) vals[j] = p[(long long)j * ldb];
            } else {
#pragma unroll
                for (int j = 0; j < 8; ++j) {
                    const int k = kk + rb + j;
                    vals[j] = (k < Kdim && n < Ndim)
                                  ? Bm[bOff + (long long)k * ldb + n] : 0.0f;
                }
            }
            const int s      = rb >> 5;
            const int r32    = rb & 31;
            const int lane_d = ((r32 & 8) << 1) + (c & 15);
            const int ebase  = (r32 & 16) >> 1;
            v8bf pk;
#pragma unroll
            for (int j = 0; j < 8; ++j) pk[j] = (bf16_t)vals[j];
            *(v8bf*)&Bsd[sbuf][c >> 4][s][lane_d][ebase] = pk;
        }
    };

    // ---- 4 WMMAs per wave per stage from LDS buffer `sbuf` ----
    auto compute = [&](int sbuf) {
#pragma unroll
        for (int s = 0; s < 2; ++s) {
            const v16bf af  = *(const v16bf*)&Asd[sbuf][wave][s][lane][0];
            const v16bf bf0 = *(const v16bf*)&Bsd[sbuf][0][s][lane][0];
            acc0 = __builtin_amdgcn_wmma_f32_16x16x32_bf16(
                       false, af, false, bf0, (short)0, acc0, false, false);
            const v16bf bf1 = *(const v16bf*)&Bsd[sbuf][1][s][lane][0];
            acc1 = __builtin_amdgcn_wmma_f32_16x16x32_bf16(
                       false, af, false, bf1, (short)0, acc1, false, false);
        }
    };

    // ---- software pipeline: prefetch next slab while computing current ----
    const int nStages = (Kdim + TK - 1) / TK;
    stage(0, 0);
    __syncthreads();
    for (int st = 0; st < nStages; ++st) {
        const int buf = st & 1;
        if (st + 1 < nStages) stage(buf ^ 1, (st + 1) * TK);
        compute(buf);
        __syncthreads();
    }

    // ---- epilogue: bias + ELU + pad-vertex zero, f32 store ----
    // C/D layout: VGPR i -> M = i + 8*(lane>>4), N = lane&15.
    const int r15 = lane & 15;
    const int h   = lane >> 4;
#pragma unroll
    for (int nt = 0; nt < 2; ++nt) {
        const int n = n0 + nt * 16 + r15;
#pragma unroll
        for (int i = 0; i < 8; ++i) {
            const int m = m0 + wave * 16 + i + 8 * h;
            if (m < M && n < Ndim) {
                float val = (nt == 0) ? acc0[i] : acc1[i];
                if (bias) val += bias[n];
                if (flags & 1) val = (val > 0.0f) ? val : expm1f(val);
                if ((flags & 2) && m == M - 1) val = 0.0f;
                C[cOff + (long long)m * ldc + n] = val;
            }
        }
    }
}

static inline void run_gemm(hipStream_t st, int batch, int M, int Kd, int Nd,
                            const float* A, long long Abs, int lda,
                            const long long* gather, const float* X, long long Xbs, int Fi,
                            const float* Bm, long long Bbs, int ldb,
                            float* C, long long Cbs, int ldc,
                            const float* bias, int flags)
{
    dim3 grid((Nd + TN - 1) / TN, (M + TM - 1) / TM, batch);
    switch (Fi) {
    case 3:
        spiral_gemm_bf16_wmma<3><<<grid, BLOCK_THREADS, 0, st>>>(
            M, Kd, Nd, A, Abs, lda, gather, X, Xbs, Bm, Bbs, ldb, C, Cbs, ldc, bias, flags);
        break;
    case 64:
        spiral_gemm_bf16_wmma<64><<<grid, BLOCK_THREADS, 0, st>>>(
            M, Kd, Nd, A, Abs, lda, gather, X, Xbs, Bm, Bbs, ldb, C, Cbs, ldc, bias, flags);
        break;
    case 128:
        spiral_gemm_bf16_wmma<128><<<grid, BLOCK_THREADS, 0, st>>>(
            M, Kd, Nd, A, Abs, lda, gather, X, Xbs, Bm, Bbs, ldb, C, Cbs, ldc, bias, flags);
        break;
    case 256:
        spiral_gemm_bf16_wmma<256><<<grid, BLOCK_THREADS, 0, st>>>(
            M, Kd, Nd, A, Abs, lda, gather, X, Xbs, Bm, Bbs, ldb, C, Cbs, ldc, bias, flags);
        break;
    default:
        spiral_gemm_bf16_wmma<0><<<grid, BLOCK_THREADS, 0, st>>>(
            M, Kd, Nd, A, Abs, lda, gather, X, Xbs, Bm, Bbs, ldb, C, Cbs, ldc, bias, flags);
        break;
    }
}

extern "C" void kernel_launch(void* const* d_in, const int* in_sizes, int n_in,
                              void* d_out, int out_size, void* d_ws, size_t ws_size,
                              hipStream_t stream) {
    (void)in_sizes; (void)n_in; (void)out_size; (void)ws_size;

    // setup_inputs() order: x, s0..s3, (D0,U0,D1,U1,D2,U2,D3,U3),
    // We0,be0..We3,be3, fceW,fceb,fcdW,fcdb, Wd0,bd0..Wd3,bd3
    const float*     x   = (const float*)d_in[0];
    const long long* s0  = (const long long*)d_in[1];
    const long long* s1  = (const long long*)d_in[2];
    const long long* s2  = (const long long*)d_in[3];
    const long long* s3  = (const long long*)d_in[4];
    const float* D0 = (const float*)d_in[5];
    const float* U0 = (const float*)d_in[6];
    const float* D1 = (const float*)d_in[7];
    const float* U1 = (const float*)d_in[8];
    const float* D2 = (const float*)d_in[9];
    const float* U2 = (const float*)d_in[10];
    const float* D3 = (const float*)d_in[11];
    const float* U3 = (const float*)d_in[12];
    const float* We0 = (const float*)d_in[13]; const float* be0 = (const float*)d_in[14];
    const float* We1 = (const float*)d_in[15]; const float* be1 = (const float*)d_in[16];
    const float* We2 = (const float*)d_in[17]; const float* be2 = (const float*)d_in[18];
    const float* We3 = (const float*)d_in[19]; const float* be3 = (const float*)d_in[20];
    const float* fceW = (const float*)d_in[21]; const float* fceb = (const float*)d_in[22];
    const float* fcdW = (const float*)d_in[23]; const float* fcdb = (const float*)d_in[24];
    const float* Wd0 = (const float*)d_in[25]; const float* bd0 = (const float*)d_in[26];
    const float* Wd1 = (const float*)d_in[27]; const float* bd1 = (const float*)d_in[28];
    const float* Wd2 = (const float*)d_in[29]; const float* bd2 = (const float*)d_in[30];
    const float* Wd3 = (const float*)d_in[31]; const float* bd3 = (const float*)d_in[32];

    const int B = 64;
    const int N0 = 5024, N1 = 1257, N2 = 315, N3 = 80, N4 = 21;

    // ping-pong f32 scratch, each sized for the largest intermediate [64,5024,64]
    const size_t BUF = (size_t)B * N0 * 64;
    float* bufA = (float*)d_ws;
    float* bufB = bufA + BUF;

    // ---------------- encoder ----------------
    run_gemm(stream, B, N0, 3 * 9, 64, nullptr, 0, 0, s0, x, (long long)N0 * 3, 3,
             We0, 0, 64, bufA, (long long)N0 * 64, 64, be0, 3);
    run_gemm(stream, B, N1, N0, 64, D0, 0, N0, nullptr, nullptr, 0, 0,
             bufA, (long long)N0 * 64, 64, bufB, (long long)N1 * 64, 64, nullptr, 0);
    run_gemm(stream, B, N1, 64 * 9, 64, nullptr, 0, 0, s1, bufB, (long long)N1 * 64, 64,
             We1, 0, 64, bufA, (long long)N1 * 64, 64, be1, 3);
    run_gemm(stream, B, N2, N1, 64, D1, 0, N1, nullptr, nullptr, 0, 0,
             bufA, (long long)N1 * 64, 64, bufB, (long long)N2 * 64, 64, nullptr, 0);
    run_gemm(stream, B, N2, 64 * 9, 128, nullptr, 0, 0, s2, bufB, (long long)N2 * 64, 64,
             We2, 0, 128, bufA, (long long)N2 * 128, 128, be2, 3);
    run_gemm(stream, B, N3, N2, 128, D2, 0, N2, nullptr, nullptr, 0, 0,
             bufA, (long long)N2 * 128, 128, bufB, (long long)N3 * 128, 128, nullptr, 0);
    run_gemm(stream, B, N3, 128 * 9, 256, nullptr, 0, 0, s3, bufB, (long long)N3 * 128, 128,
             We3, 0, 256, bufA, (long long)N3 * 256, 256, be3, 3);
    run_gemm(stream, B, N4, N3, 256, D3, 0, N3, nullptr, nullptr, 0, 0,
             bufA, (long long)N3 * 256, 256, bufB, (long long)N4 * 256, 256, nullptr, 0);

    // ---------------- bottleneck ----------------
    run_gemm(stream, 1, B, N4 * 256, 256, bufB, 0, N4 * 256, nullptr, nullptr, 0, 0,
             fceW, 0, 256, bufA, 0, 256, fceb, 0);
    run_gemm(stream, 1, B, 256, N4 * 256, bufA, 0, 256, nullptr, nullptr, 0, 0,
             fcdW, 0, N4 * 256, bufB, 0, N4 * 256, fcdb, 0);

    // ---------------- decoder ----------------
    run_gemm(stream, B, N3, N4, 256, U3, 0, N4, nullptr, nullptr, 0, 0,
             bufB, (long long)N4 * 256, 256, bufA, (long long)N3 * 256, 256, nullptr, 0);
    run_gemm(stream, B, N3, 256 * 9, 128, nullptr, 0, 0, s3, bufA, (long long)N3 * 256, 256,
             Wd0, 0, 128, bufB, (long long)N3 * 128, 128, bd0, 3);
    run_gemm(stream, B, N2, N3, 128, U2, 0, N3, nullptr, nullptr, 0, 0,
             bufB, (long long)N3 * 128, 128, bufA, (long long)N2 * 128, 128, nullptr, 0);
    run_gemm(stream, B, N2, 128 * 9, 64, nullptr, 0, 0, s2, bufA, (long long)N2 * 128, 128,
             Wd1, 0, 64, bufB, (long long)N2 * 64, 64, bd1, 3);
    run_gemm(stream, B, N1, N2, 64, U1, 0, N2, nullptr, nullptr, 0, 0,
             bufB, (long long)N2 * 64, 64, bufA, (long long)N1 * 64, 64, nullptr, 0);
    run_gemm(stream, B, N1, 64 * 9, 64, nullptr, 0, 0, s1, bufA, (long long)N1 * 64, 64,
             Wd2, 0, 64, bufB, (long long)N1 * 64, 64, bd2, 3);
    run_gemm(stream, B, N0, N1, 64, U0, 0, N1, nullptr, nullptr, 0, 0,
             bufB, (long long)N1 * 64, 64, bufA, (long long)N0 * 64, 64, nullptr, 0);
    run_gemm(stream, B, N0, 64 * 9, 3, nullptr, 0, 0, s0, bufA, (long long)N0 * 64, 64,
             Wd3, 0, 3, (float*)d_out, (long long)N0 * 3, 3, bd3, 2);
}
